// Transformer_8967891714177
// MI455X (gfx1250) — matrix operand order repaired and checked by reading, not compiled, not run on hardware
//
#include <hip/hip_runtime.h>
#include <cstdint>
#include <cstddef>

typedef __attribute__((ext_vector_type(16))) __bf16 v16bf;
typedef __attribute__((ext_vector_type(8)))  __bf16 v8bf;
typedef __attribute__((ext_vector_type(8)))  float  v8f;

namespace {
constexpr int kB    = 2;
constexpr int kLTOT = 24480;           // 18432 + 4608 + 1152 + 288
constexpr int kM    = kB * kLTOT;      // 48960 token rows
constexpr int kD    = 256;
constexpr int kDFF  = 1024;
}

// ---------- weight convert + transpose: W[K,N] f32 -> WT[N,K] bf16 ----------
__global__ void k_wT(const float* __restrict__ W, __bf16* __restrict__ WT,
                     int K, int N) {
  int idx = blockIdx.x * 256 + threadIdx.x;
  if (idx >= K * N) return;
  int k = idx / N, n = idx - k * N;
  WT[(size_t)n * K + k] = (__bf16)W[idx];
}

// ---------- q = cur+pos -> bf16 ; x = cur -> bf16 ----------
__global__ void k_prep(const float* __restrict__ cur, const float* __restrict__ pos,
                       __bf16* __restrict__ qb, __bf16* __restrict__ xb, int n) {
  int i = blockIdx.x * 256 + threadIdx.x;
  if (i >= n) return;
  float c = cur[i];
  qb[i] = (__bf16)(c + pos[i]);
  xb[i] = (__bf16)c;
}

// ---------- WMMA GEMM: out[M,N] = A[M,K]bf16 * BT[N,K]^T + bias ----------
// Wave computes a 32x64 tile: 2 A fragments x 4 B fragments -> 8 accumulators,
// so each A frag is reused 4x and each B frag 2x in-register (12 b128 loads per
// 8 v_wmma_f32_16x16x32_bf16). Block = 4 waves arranged 2(M) x 2(N): 64x128 tile.
// EPI: 0 = f32 out, 1 = bf16 out, 2 = relu->bf16, 3 = resid + f32 out.
template<int EPI>
__global__ __launch_bounds__(128) void k_gemm(
    const __bf16* __restrict__ A, const __bf16* __restrict__ BT,
    const float* __restrict__ bias, const float* __restrict__ resid,
    float* __restrict__ outf, __bf16* __restrict__ outb,
    int N, int K) {
  const int wv   = threadIdx.x >> 5;
  const int lane = threadIdx.x & 31;
  const int hf   = lane >> 4;     // which half-wave
  const int lr   = lane & 15;     // A: M row / B,C,D: N column
  const int wm   = wv & 1;        // wave's M position in block
  const int wn   = wv >> 1;       // wave's N position in block
  const int row0 = blockIdx.y * 64 + wm * 32;
  const int n0   = blockIdx.x * 128 + wn * 64;

  // A 16x32 bf16 layout: lane holds K = hf*16 + [0..15]
  const __bf16* Arow0 = A + (size_t)(row0 + lr) * K + hf * 16;
  const __bf16* Arow1 = Arow0 + (size_t)16 * K;
  // B 32x16 bf16 layout: lane holds 16 contiguous K (hf selects K block) of col lr
  const __bf16* Bcol = BT + (size_t)(n0 + lr) * K + hf * 16;

  v8f acc[2][4] = {};
  for (int k0 = 0; k0 < K; k0 += 32) {
    v8bf a0lo = *(const v8bf*)(Arow0 + k0);        // 16B aligned
    v8bf a0hi = *(const v8bf*)(Arow0 + k0 + 8);
    v8bf a1lo = *(const v8bf*)(Arow1 + k0);
    v8bf a1hi = *(const v8bf*)(Arow1 + k0 + 8);
    v16bf a0 = __builtin_shufflevector(a0lo, a0hi,
        0,1,2,3,4,5,6,7,8,9,10,11,12,13,14,15);
    v16bf a1 = __builtin_shufflevector(a1lo, a1hi,
        0,1,2,3,4,5,6,7,8,9,10,11,12,13,14,15);
#pragma unroll
    for (int t = 0; t < 4; ++t) {
      v16bf bfrag = *(const v16bf*)(Bcol + (size_t)t * 16 * K + k0); // 32B aligned
      acc[0][t] = __builtin_amdgcn_wmma_f32_16x16x32_bf16(
          false, a0, false, bfrag, (short)0, acc[0][t], false, false);
      acc[1][t] = __builtin_amdgcn_wmma_f32_16x16x32_bf16(
          false, a1, false, bfrag, (short)0, acc[1][t], false, false);
    }
  }
  // D layout: VGPR r holds row = tile_row0 + hf*8 + r, column = n0 + t*16 + lr
#pragma unroll
  for (int m = 0; m < 2; ++m) {
#pragma unroll
    for (int t = 0; t < 4; ++t) {
      int col = n0 + t * 16 + lr;
      float bia = bias[col];
#pragma unroll
      for (int r = 0; r < 8; ++r) {
        int row = row0 + m * 16 + hf * 8 + r;
        size_t o = (size_t)row * N + col;
        float v = acc[m][t][r] + bia;
        if (EPI == 0)      outf[o] = v;
        else if (EPI == 1) outb[o] = (__bf16)v;
        else if (EPI == 2) outb[o] = (__bf16)fmaxf(v, 0.f);
        else               outf[o] = resid[o] + v;
      }
    }
  }
}

// ---------- deformable 3D sampling ----------
// Block = one (b, query): 8 heads x 32 channels. Wave = one head, lane = channel
// so the 8-corner gathers are 64B-contiguous per wave (val is L2-resident).
__global__ __launch_bounds__(256) void k_sample(
    const float* __restrict__ off,   // [M,384] = (h, lvl, p, xyz)
    const float* __restrict__ awl,   // [M,128] = (h, lvl*4+p) logits
    const __bf16* __restrict__ valb, // [M,256]
    const float* __restrict__ vr,    // [B,4,3] valid_ratios (x,y,t)
    __bf16* __restrict__ attnb) {    // [M,256]
  constexpr int TT[4] = {8, 8, 8, 8};
  constexpr int HH[4] = {48, 24, 12, 6};
  constexpr int WW[4] = {48, 24, 12, 6};
  constexpr int LS[4] = {0, 18432, 23040, 24192};

  const int l = blockIdx.x;
  const int b = blockIdx.y;
  const int h = threadIdx.x >> 5;
  const int d = threadIdx.x & 31;
  const size_t row = (size_t)b * kLTOT + l;

  const int lvlq = (l < 18432) ? 0 : (l < 23040 ? 1 : (l < 24192 ? 2 : 3));
  const int rem = l - LS[lvlq];
  const int Wq = WW[lvlq], Hq = HH[lvlq], Tq = TT[lvlq];
  const int xq = rem % Wq;
  const int yq = (rem / Wq) % Hq;
  const int tq = rem / (Wq * Hq);
  const float* vrb = vr + b * 12;
  const float bx = (xq + 0.5f) / (vrb[lvlq * 3 + 0] * Wq);
  const float by = (yq + 0.5f) / (vrb[lvlq * 3 + 1] * Hq);
  const float bt = (tq + 0.5f) / (vrb[lvlq * 3 + 2] * Tq);

  // softmax over this head's 16 (level, point) logits
  const float* lg = awl + row * 128 + h * 16;
  float mx = lg[0];
#pragma unroll
  for (int j = 1; j < 16; ++j) mx = fmaxf(mx, lg[j]);
  float e[16];
  float s = 0.f;
#pragma unroll
  for (int j = 0; j < 16; ++j) { e[j] = __expf(lg[j] - mx); s += e[j]; }
  const float inv = 1.f / s;

  const float* op = off + row * 384 + h * 48;
  float acc = 0.f;
#pragma unroll
  for (int lvl = 0; lvl < 4; ++lvl) {
    const int T = TT[lvl], H = HH[lvl], W = WW[lvl];
    const size_t base = (size_t)b * kLTOT + LS[lvl];
    const float rx = bx * vrb[lvl * 3 + 0] * W;
    const float ry = by * vrb[lvl * 3 + 1] * H;
    const float rt = bt * vrb[lvl * 3 + 2] * T;
#pragma unroll
    for (int p = 0; p < 4; ++p) {
      const float* o = op + (lvl * 4 + p) * 3;
      // loc = ref + off/norm ; x = loc*W - 0.5  ->  x = ref*W + off - 0.5
      const float x = rx + o[0] - 0.5f;
      const float y = ry + o[1] - 0.5f;
      const float t = rt + o[2] - 0.5f;
      const float xf = floorf(x), yf = floorf(y), tf = floorf(t);
      const float fx = x - xf, fy = y - yf, ft = t - tf;
      const int x0 = (int)xf, y0 = (int)yf, t0 = (int)tf;
      const float aw = e[lvl * 4 + p] * inv;
      float samp = 0.f;
#pragma unroll
      for (int dt = 0; dt < 2; ++dt) {
        const int ti = t0 + dt;
        if (ti < 0 || ti >= T) continue;
        const float wt = dt ? ft : 1.f - ft;
#pragma unroll
        for (int dy = 0; dy < 2; ++dy) {
          const int yi = y0 + dy;
          if (yi < 0 || yi >= H) continue;
          const float wy = dy ? fy : 1.f - fy;
#pragma unroll
          for (int dx = 0; dx < 2; ++dx) {
            const int xi = x0 + dx;
            if (xi < 0 || xi >= W) continue;
            const float wx = dx ? fx : 1.f - fx;
            const size_t idx = base + ((size_t)ti * H + yi) * W + xi;
            samp += wt * wy * wx * (float)valb[idx * 256 + h * 32 + d];
          }
        }
      }
      acc += aw * samp;
    }
  }
  attnb[row * 256 + h * 32 + d] = (__bf16)acc;
}

// ---------- LayerNorm: wave per 256-wide row, 8 rows per block ----------
__global__ __launch_bounds__(256) void k_ln(
    const float* __restrict__ x, const float* __restrict__ g,
    const float* __restrict__ bta, float* __restrict__ out,
    __bf16* __restrict__ outb) {
  const int wv = threadIdx.x >> 5;
  const int lane = threadIdx.x & 31;
  const size_t row = (size_t)blockIdx.x * 8 + wv;
  const float* xr = x + row * 256;
  float v[8];
  float s = 0.f, sq = 0.f;
#pragma unroll
  for (int j = 0; j < 8; ++j) {
    v[j] = xr[lane + 32 * j];
    s += v[j]; sq += v[j] * v[j];
  }
#pragma unroll
  for (int m = 16; m >= 1; m >>= 1) {
    s  += __shfl_xor(s, m, 32);
    sq += __shfl_xor(sq, m, 32);
  }
  const float mean = s * (1.f / 256.f);
  const float var  = sq * (1.f / 256.f) - mean * mean;
  const float rs   = rsqrtf(var + 1e-5f);
#pragma unroll
  for (int j = 0; j < 8; ++j) {
    const int c = lane + 32 * j;
    const float y = (v[j] - mean) * rs * g[c] + bta[c];
    out[row * 256 + c]  = y;
    outb[row * 256 + c] = (__bf16)y;
  }
}

extern "C" void kernel_launch(void* const* d_in, const int* in_sizes, int n_in,
                              void* d_out, int out_size, void* d_ws, size_t ws_size,
                              hipStream_t stream) {
  const float* src  = (const float*)d_in[0];
  const float* pos  = (const float*)d_in[1];
  const float* vr   = (const float*)d_in[2];
  const float* Wo   = (const float*)d_in[5];
  const float* bo   = (const float*)d_in[6];
  const float* Wa   = (const float*)d_in[7];
  const float* ba   = (const float*)d_in[8];
  const float* Wv   = (const float*)d_in[9];
  const float* bv   = (const float*)d_in[10];
  const float* Wout = (const float*)d_in[11];
  const float* bout = (const float*)d_in[12];
  const float* g1   = (const float*)d_in[13];
  const float* b1   = (const float*)d_in[14];
  const float* W1   = (const float*)d_in[15];
  const float* bl1  = (const float*)d_in[16];
  const float* W2   = (const float*)d_in[17];
  const float* bl2  = (const float*)d_in[18];
  const float* g2   = (const float*)d_in[19];
  const float* b2   = (const float*)d_in[20];
  float* cur = (float*)d_out;   // running fp32 state lives in d_out

  char* p = (char*)d_ws;
  auto carve = [&](size_t bytes) -> void* {
    void* r = (void*)p;
    p += (bytes + 255) & ~(size_t)255;
    return r;
  };
  __bf16* qb    = (__bf16*)carve((size_t)kM * kD * 2);
  __bf16* xb    = (__bf16*)carve((size_t)kM * kD * 2);
  __bf16* valb  = (__bf16*)carve((size_t)kM * kD * 2);
  __bf16* attnb = (__bf16*)carve((size_t)kM * kD * 2);
  __bf16* hb    = (__bf16*)carve((size_t)kM * kDFF * 2);
  float*  offb  = (float*)carve((size_t)kM * 384 * 4);  // also residual tmp [M,256]
  float*  awlb  = (float*)carve((size_t)kM * 128 * 4);
  __bf16* WvT   = (__bf16*)carve((size_t)256 * 256 * 2);
  __bf16* WoT   = (__bf16*)carve((size_t)384 * 256 * 2);
  __bf16* WaT   = (__bf16*)carve((size_t)128 * 256 * 2);
  __bf16* WouT  = (__bf16*)carve((size_t)256 * 256 * 2);
  __bf16* W1T   = (__bf16*)carve((size_t)1024 * 256 * 2);
  __bf16* W2T   = (__bf16*)carve((size_t)256 * 1024 * 2);

  hipMemcpyAsync(cur, src, (size_t)kM * kD * 4, hipMemcpyDeviceToDevice, stream);

  const int nElem = kM * kD;
  for (int i = 0; i < 2; ++i) {
    k_wT<<<(256 * 384 + 255) / 256, 256, 0, stream>>>(Wo   + (size_t)i * 256 * 384, WoT, 256, 384);
    k_wT<<<(256 * 128 + 255) / 256, 256, 0, stream>>>(Wa   + (size_t)i * 256 * 128, WaT, 256, 128);
    k_wT<<<(256 * 256 + 255) / 256, 256, 0, stream>>>(Wv   + (size_t)i * 256 * 256, WvT, 256, 256);
    k_wT<<<(256 * 256 + 255) / 256, 256, 0, stream>>>(Wout + (size_t)i * 256 * 256, WouT, 256, 256);
    k_wT<<<(256 * 1024 + 255) / 256, 256, 0, stream>>>(W1  + (size_t)i * 256 * 1024, W1T, 256, 1024);
    k_wT<<<(1024 * 256 + 255) / 256, 256, 0, stream>>>(W2  + (size_t)i * 1024 * 256, W2T, 1024, 256);

    k_prep<<<nElem / 256, 256, 0, stream>>>(cur, pos, qb, xb, nElem);

    // val = x @ Wv + bv  -> bf16
    k_gemm<1><<<dim3(256 / 128, kM / 64), 128, 0, stream>>>(xb, WvT, bv + (size_t)i * 256, nullptr, nullptr, valb, 256, 256);
    // off = q @ Wo + bo  -> f32
    k_gemm<0><<<dim3(384 / 128, kM / 64), 128, 0, stream>>>(qb, WoT, bo + (size_t)i * 384, nullptr, offb, nullptr, 384, 256);
    // attention logits = q @ Wa + ba -> f32
    k_gemm<0><<<dim3(128 / 128, kM / 64), 128, 0, stream>>>(qb, WaT, ba + (size_t)i * 128, nullptr, awlb, nullptr, 128, 256);

    k_sample<<<dim3(kLTOT, kB), 256, 0, stream>>>(offb, awlb, valb, vr, attnb);

    // tmp = cur + attn @ Wout + bout  (offb reused as tmp[M,256])
    k_gemm<3><<<dim3(256 / 128, kM / 64), 128, 0, stream>>>(attnb, WouT, bout + (size_t)i * 256, cur, offb, nullptr, 256, 256);
    k_ln<<<kM / 8, 256, 0, stream>>>(offb, g1 + (size_t)i * 256, b1 + (size_t)i * 256, cur, xb);

    // h = relu(x @ W1 + bl1) -> bf16
    k_gemm<2><<<dim3(1024 / 128, kM / 64), 128, 0, stream>>>(xb, W1T, bl1 + (size_t)i * 1024, nullptr, nullptr, hb, 1024, 256);
    // tmp = cur + h @ W2 + bl2
    k_gemm<3><<<dim3(256 / 128, kM / 64), 128, 0, stream>>>(hb, W2T, bl2 + (size_t)i * 256, cur, offb, nullptr, 256, 1024);
    k_ln<<<kM / 8, 256, 0, stream>>>(offb, g2 + (size_t)i * 256, b2 + (size_t)i * 256, cur, xb);
  }
  (void)in_sizes; (void)n_in; (void)out_size; (void)ws_size;
}
